// LSTMpred2_27496380629543
// MI455X (gfx1250) — compile-verified, matrix operand order checked by reading, not run on hardware
//
#include <hip/hip_runtime.h>
#include <hip/hip_bf16.h>

// ---------------------------------------------------------------------------
// LSTM seq2seq (H=512, B=1024, T=64, NL=3) for gfx1250 (MI455X).
// Per-cell GEMM: M=1024 (batch), N=2048 (4 gates), K=1024 ([x;h]) in bf16 WMMA
// with fused gate nonlinearity + cell update epilogue (gate-grouped N tiling).
// Staging uses GLOBAL_LOAD_ASYNC_TO_LDS (ASYNCcnt) with double-buffered tiles.
// ---------------------------------------------------------------------------

typedef __bf16 bf16x16 __attribute__((ext_vector_type(16)));
typedef float  f32x8   __attribute__((ext_vector_type(8)));
typedef int    v4i_gcc __attribute__((vector_size(16)));   // matches builtin proto

#define HID   512
#define BATCH 1024
#define BHE   (1024 * 512)        // elements of one (B,H) activation matrix
#define NLAY  3
#define TENC  63
#define TDEC  64

union FragU { uint4 u[2]; bf16x16 v; };

__device__ __forceinline__ float sigmoidf_(float x) {
    return 1.0f / (1.0f + __expf(-x));
}

// ---- async global -> LDS copy (CDNA5), with sync fallback --------------
#if __has_builtin(__builtin_amdgcn_global_load_async_to_lds_b128)
#define HAVE_ASYNC_LDS 1
#else
#define HAVE_ASYNC_LDS 0
#endif

__device__ __forceinline__ void copy16_to_lds(const __hip_bfloat16* g, unsigned short* l)
{
#if HAVE_ASYNC_LDS
    __builtin_amdgcn_global_load_async_to_lds_b128(
        (__attribute__((address_space(1))) v4i_gcc*)g,
        (__attribute__((address_space(3))) v4i_gcc*)l, 0, 0);
#else
    *reinterpret_cast<uint4*>(l) = *reinterpret_cast<const uint4*>(g);
#endif
}

__device__ __forceinline__ void wait_lds_copies()
{
#if HAVE_ASYNC_LDS
#if __has_builtin(__builtin_amdgcn_s_wait_asynccnt)
    __builtin_amdgcn_s_wait_asynccnt(0);
#else
    asm volatile("s_wait_asynccnt 0" ::: "memory");
#endif
#endif
}

// ---- shared tile helpers ------------------------------------------------
__device__ __forceinline__ void mma_tile(const unsigned short* sA, const unsigned short* sB,
                                         int wm, int wn, int lh, int l15, f32x8 acc[4][2])
{
#pragma unroll
    for (int kk = 0; kk < 64; kk += 32) {
        const int ks = kk + 8 * lh;
        FragU a[4], b[2];
#pragma unroll
        for (int fi = 0; fi < 4; ++fi) {
            int mr = wm * 64 + fi * 16 + l15;
            a[fi].u[0] = *reinterpret_cast<const uint4*>(&sA[mr * 64 + ks]);
            a[fi].u[1] = *reinterpret_cast<const uint4*>(&sA[mr * 64 + ks + 16]);
        }
#pragma unroll
        for (int fj = 0; fj < 2; ++fj) {
            int nr = wn * 32 + fj * 16 + l15;
            b[fj].u[0] = *reinterpret_cast<const uint4*>(&sB[nr * 64 + ks]);
            b[fj].u[1] = *reinterpret_cast<const uint4*>(&sB[nr * 64 + ks + 16]);
        }
#pragma unroll
        for (int fi = 0; fi < 4; ++fi)
#pragma unroll
            for (int fj = 0; fj < 2; ++fj)
                acc[fi][fj] = __builtin_amdgcn_wmma_f32_16x16x32_bf16(
                    false, a[fi].v, false, b[fj].v, (short)0, acc[fi][fj], false, false);
    }
}

// ---------------------------------------------------------------------------
// Fused LSTM cell: gates = [Ax | Ah] * W^T  (W is 2048 x 1024 row-major),
// then i/f/g/o nonlinearity, c update (in place), h stored as bf16.
// Block tile: 128 (batch) x 128 (4 gates x 32 hidden cols). 8 waves,
// wave grid 2(M) x 4(N); wave-N index == gate index.
// ---------------------------------------------------------------------------
__device__ __forceinline__ void stage_cell_tile(
    const __hip_bfloat16* __restrict__ Ax, const __hip_bfloat16* __restrict__ Ah,
    const __hip_bfloat16* __restrict__ W, int kt, int m0, int j0, int tid,
    unsigned short* sA, unsigned short* sB)
{
    const int ck = (kt & 7) * 64;                   // col inside 512-wide half
    const __hip_bfloat16* Asrc = (kt < 8) ? Ax : Ah;
#pragma unroll
    for (int i = 0; i < 4; ++i) {                   // A tile: 128 x 64 bf16
        int q   = tid + 256 * i;
        int row = q >> 3;
        int ch  = (q & 7) * 8;
        copy16_to_lds(Asrc + (size_t)(m0 + row) * HID + ck + ch, &sA[row * 64 + ch]);
    }
#pragma unroll
    for (int i = 0; i < 4; ++i) {                   // B tile: gate*32+r -> W row gate*512+j0+r
        int q    = tid + 256 * i;
        int row  = q >> 3;
        int ch   = (q & 7) * 8;
        int wrow = (row >> 5) * HID + j0 + (row & 31);
        copy16_to_lds(W + (size_t)wrow * 1024 + kt * 64 + ch, &sB[row * 64 + ch]);
    }
}

__global__ __launch_bounds__(256)
void lstm_cell_kernel(const __hip_bfloat16* __restrict__ Ax,   // B x 512 bf16
                      const __hip_bfloat16* __restrict__ Ah,   // B x 512 bf16
                      const __hip_bfloat16* __restrict__ W,    // 2048 x 1024 bf16
                      const float* __restrict__ bias,          // 2048 (bih+bhh)
                      float* __restrict__ Cst,                 // B x 512 f32 (in place)
                      __hip_bfloat16* __restrict__ Hd0,        // B x 512 bf16
                      __hip_bfloat16* Hd1)                     // optional extra dest
{
    // 64 KB: two double-buffered (A,B) tile pairs; sG aliases buffer 1,
    // which is only live again after the K loop's final barrier.
    __shared__ __align__(16) unsigned char smem[65536];
    unsigned short* sA0 = (unsigned short*)(smem);
    unsigned short* sB0 = (unsigned short*)(smem + 16384);
    unsigned short* sA1 = (unsigned short*)(smem + 32768);
    unsigned short* sB1 = (unsigned short*)(smem + 49152);
    float*          sG  = (float*)(smem + 32768);   // 64 x 128 f32 gate exchange

    const int tid  = threadIdx.x;
    const int lane = tid & 31;
    const int wid  = tid >> 5;
    const int wm   = wid >> 2;      // 0..1 : M half
    const int wn   = wid & 3;       // 0..3 : gate index
    const int m0   = blockIdx.x * 128;
    const int j0   = blockIdx.y * 32;
    const int lh   = lane >> 4;
    const int l15  = lane & 15;

    f32x8 acc[4][2];
    const f32x8 zero = {0.f, 0.f, 0.f, 0.f, 0.f, 0.f, 0.f, 0.f};
#pragma unroll
    for (int i = 0; i < 4; ++i)
#pragma unroll
        for (int j = 0; j < 2; ++j) acc[i][j] = zero;

    stage_cell_tile(Ax, Ah, W, 0, m0, j0, tid, sA0, sB0);
    wait_lds_copies();
    __syncthreads();

    for (int kt = 0; kt < 16; ++kt) {               // K = 1024, BK = 64
        unsigned short* cA = (kt & 1) ? sA1 : sA0;
        unsigned short* cB = (kt & 1) ? sB1 : sB0;
        if (kt + 1 < 16) {
            unsigned short* nA = (kt & 1) ? sA0 : sA1;
            unsigned short* nB = (kt & 1) ? sB0 : sB1;
            stage_cell_tile(Ax, Ah, W, kt + 1, m0, j0, tid, nA, nB);
        }
        mma_tile(cA, cB, wm, wn, lh, l15, acc);
        wait_lds_copies();
        __syncthreads();
    }

    // ------------------------------------------------------------------
    // Epilogue: exchange gate tiles through LDS (two 64-row passes) and
    // apply the LSTM cell update fully fused.
    // ------------------------------------------------------------------
#pragma unroll
    for (int half = 0; half < 2; ++half) {
        if (wm == half) {
#pragma unroll
            for (int fi = 0; fi < 4; ++fi)
#pragma unroll
                for (int fj = 0; fj < 2; ++fj)
#pragma unroll
                    for (int r = 0; r < 8; ++r) {
                        int lr = fi * 16 + r + 8 * lh;           // 0..63
                        int lc = wn * 32 + fj * 16 + l15;        // 0..127
                        sG[lr * 128 + lc] = acc[fi][fj][r];
                    }
        }
        __syncthreads();
        {
            const int jc = tid & 31;
            const int mb = tid >> 5;                 // 0..7
            const int gj = j0 + jc;
            const float bI = bias[0 * HID + gj];
            const float bF = bias[1 * HID + gj];
            const float bG = bias[2 * HID + gj];
            const float bO = bias[3 * HID + gj];
#pragma unroll
            for (int r = 0; r < 8; ++r) {
                int lm = mb + 8 * r;
                int gm = m0 + half * 64 + lm;
                float gi = sG[lm * 128 +      jc] + bI;
                float gf = sG[lm * 128 + 32 + jc] + bF;
                float gg = sG[lm * 128 + 64 + jc] + bG;
                float go = sG[lm * 128 + 96 + jc] + bO;
                float cv = Cst[(size_t)gm * HID + gj];
                float c2 = sigmoidf_(gf) * cv + sigmoidf_(gi) * tanhf(gg);
                float h2 = sigmoidf_(go) * tanhf(c2);
                Cst[(size_t)gm * HID + gj] = c2;
                __hip_bfloat16 hb = __float2bfloat16(h2);
                Hd0[(size_t)gm * HID + gj] = hb;
                if (Hd1) Hd1[(size_t)gm * HID + gj] = hb;
            }
        }
        __syncthreads();
    }
}

// ---------------------------------------------------------------------------
// Output projection for one decoder step:
// Out[b, o, step] = sum_i W[o,i] * Ain[b,i] + bias[o].
// M=1024, N=512, K=512.  Block 128x128, 8 waves, same wave layout.
// ---------------------------------------------------------------------------
__device__ __forceinline__ void stage_proj_tile(
    const __hip_bfloat16* __restrict__ Ain, const __hip_bfloat16* __restrict__ W,
    int kt, int m0, int n0, int tid, unsigned short* sA, unsigned short* sB)
{
    const int ck = kt * 64;
#pragma unroll
    for (int i = 0; i < 4; ++i) {
        int q   = tid + 256 * i;
        int row = q >> 3;
        int ch  = (q & 7) * 8;
        copy16_to_lds(Ain + (size_t)(m0 + row) * HID + ck + ch, &sA[row * 64 + ch]);
        copy16_to_lds(W   + (size_t)(n0 + row) * HID + ck + ch, &sB[row * 64 + ch]);
    }
}

__global__ __launch_bounds__(256)
void proj_kernel(const __hip_bfloat16* __restrict__ Ain,  // B x 512 bf16
                 const __hip_bfloat16* __restrict__ W,    // 512 x 512 bf16
                 const float* __restrict__ bias,          // 512 f32
                 float* __restrict__ Out,                 // B x 512 x 64 f32
                 int step)
{
    __shared__ __align__(16) unsigned char smem[65536];
    unsigned short* sA0 = (unsigned short*)(smem);
    unsigned short* sB0 = (unsigned short*)(smem + 16384);
    unsigned short* sA1 = (unsigned short*)(smem + 32768);
    unsigned short* sB1 = (unsigned short*)(smem + 49152);

    const int tid  = threadIdx.x;
    const int lane = tid & 31;
    const int wid  = tid >> 5;
    const int wm   = wid >> 2;
    const int wn   = wid & 3;
    const int m0   = blockIdx.x * 128;
    const int n0   = blockIdx.y * 128;
    const int lh   = lane >> 4;
    const int l15  = lane & 15;

    f32x8 acc[4][2];
    const f32x8 zero = {0.f, 0.f, 0.f, 0.f, 0.f, 0.f, 0.f, 0.f};
#pragma unroll
    for (int i = 0; i < 4; ++i)
#pragma unroll
        for (int j = 0; j < 2; ++j) acc[i][j] = zero;

    stage_proj_tile(Ain, W, 0, m0, n0, tid, sA0, sB0);
    wait_lds_copies();
    __syncthreads();

    for (int kt = 0; kt < 8; ++kt) {               // K = 512, BK = 64
        unsigned short* cA = (kt & 1) ? sA1 : sA0;
        unsigned short* cB = (kt & 1) ? sB1 : sB0;
        if (kt + 1 < 8) {
            unsigned short* nA = (kt & 1) ? sA0 : sA1;
            unsigned short* nB = (kt & 1) ? sB0 : sB1;
            stage_proj_tile(Ain, W, kt + 1, m0, n0, tid, nA, nB);
        }
        mma_tile(cA, cB, wm, wn, lh, l15, acc);
        wait_lds_copies();
        __syncthreads();
    }

    // direct register scatter: C layout M = r + 8*(lane>=16), N = lane%16
#pragma unroll
    for (int fi = 0; fi < 4; ++fi)
#pragma unroll
        for (int fj = 0; fj < 2; ++fj)
#pragma unroll
            for (int r = 0; r < 8; ++r) {
                int gm = m0 + wm * 64 + fi * 16 + r + 8 * lh;
                int gn = n0 + wn * 32 + fj * 16 + l15;
                Out[((size_t)gm * HID + gn) * 64 + step] = acc[fi][fj][r] + bias[gn];
            }
}

// ---------------------------------------------------------------------------
// Prep / utility kernels
// ---------------------------------------------------------------------------
__global__ void prep_wcat(const float* __restrict__ Wih, const float* __restrict__ Whh,
                          __hip_bfloat16* __restrict__ Wcat, int total)
{
    int e = blockIdx.x * blockDim.x + threadIdx.x;
    if (e >= total) return;
    int k  = e & 1023;
    int ln = e >> 10;           // l*2048 + n
    float v = (k < 512) ? Wih[(size_t)ln * 512 + k] : Whh[(size_t)ln * 512 + (k - 512)];
    Wcat[e] = __float2bfloat16(v);
}

__global__ void prep_bias(const float* __restrict__ a, const float* __restrict__ b,
                          float* __restrict__ o, int n)
{
    int e = blockIdx.x * blockDim.x + threadIdx.x;
    if (e < n) o[e] = a[e] + b[e];
}

__global__ void prep_outw(const float* __restrict__ w, __hip_bfloat16* __restrict__ o, int n)
{
    int e = blockIdx.x * blockDim.x + threadIdx.x;
    if (e < n) o[e] = __float2bfloat16(w[e]);
}

// seq (B, H, T) f32 -> Xbf[t](B, H) bf16, t = 0..62
__global__ void xpose_kernel(const float* __restrict__ seq, __hip_bfloat16* __restrict__ Xbf)
{
    int bh = blockIdx.x * blockDim.x + threadIdx.x;
    if (bh >= BHE) return;
    for (int t = 0; t < TENC; ++t)
        Xbf[(size_t)t * BHE + bh] = __float2bfloat16(seq[(size_t)bh * 64 + t]);
}

__global__ void zero_kernel(uint4* __restrict__ p, size_t n16)
{
    const uint4 z = {0u, 0u, 0u, 0u};
    for (size_t i = blockIdx.x * blockDim.x + threadIdx.x; i < n16;
         i += (size_t)gridDim.x * blockDim.x)
        p[i] = z;
}

// ---------------------------------------------------------------------------
extern "C" void kernel_launch(void* const* d_in, const int* in_sizes, int n_in,
                              void* d_out, int out_size, void* d_ws, size_t ws_size,
                              hipStream_t stream)
{
    (void)in_sizes; (void)n_in; (void)out_size; (void)ws_size;

    const float* seq  = (const float*)d_in[0];
    const float* eWih = (const float*)d_in[1];
    const float* eWhh = (const float*)d_in[2];
    const float* eBih = (const float*)d_in[3];
    const float* eBhh = (const float*)d_in[4];
    const float* dWih = (const float*)d_in[5];
    const float* dWhh = (const float*)d_in[6];
    const float* dBih = (const float*)d_in[7];
    const float* dBhh = (const float*)d_in[8];
    const float* oW   = (const float*)d_in[9];
    const float* oB   = (const float*)d_in[10];
    float* out = (float*)d_out;                    // (B, H, 64) f32

    char* ws = (char*)d_ws;
    // workspace layout (bytes)
    const size_t szWcat = (size_t)NLAY * 2048 * 1024 * 2;   // 12.58 MB each
    size_t off = 0;
    __hip_bfloat16* WcatE = (__hip_bfloat16*)(ws + off); off += szWcat;
    __hip_bfloat16* WcatD = (__hip_bfloat16*)(ws + off); off += szWcat;
    float* bE = (float*)(ws + off); off += (size_t)NLAY * 2048 * 4;
    float* bD = (float*)(ws + off); off += (size_t)NLAY * 2048 * 4;
    __hip_bfloat16* OutWbf = (__hip_bfloat16*)(ws + off); off += (size_t)512 * 512 * 2;
    __hip_bfloat16* Xbf = (__hip_bfloat16*)(ws + off); off += (size_t)TENC * BHE * 2;
    // zero-initialized state region starts here
    size_t zoff = off;
    __hip_bfloat16* Hbuf = (__hip_bfloat16*)(ws + off); off += (size_t)NLAY * 2 * BHE * 2;
    __hip_bfloat16* Din  = (__hip_bfloat16*)(ws + off); off += (size_t)2 * BHE * 2;
    float* Cbuf = (float*)(ws + off); off += (size_t)NLAY * BHE * 4;
    size_t zbytes = off - zoff;

    // ---- prep ----
    const int wtot = NLAY * 2048 * 1024;
    prep_wcat<<<(wtot + 255) / 256, 256, 0, stream>>>(eWih, eWhh, WcatE, wtot);
    prep_wcat<<<(wtot + 255) / 256, 256, 0, stream>>>(dWih, dWhh, WcatD, wtot);
    prep_bias<<<(NLAY * 2048 + 255) / 256, 256, 0, stream>>>(eBih, eBhh, bE, NLAY * 2048);
    prep_bias<<<(NLAY * 2048 + 255) / 256, 256, 0, stream>>>(dBih, dBhh, bD, NLAY * 2048);
    prep_outw<<<(512 * 512 + 255) / 256, 256, 0, stream>>>(oW, OutWbf, 512 * 512);
    xpose_kernel<<<(BHE + 255) / 256, 256, 0, stream>>>(seq, Xbf);
    zero_kernel<<<1024, 256, 0, stream>>>((uint4*)(ws + zoff), zbytes / 16);

    const dim3 cgrid(8, 16);   // 1024/128 x 512/32
    const dim3 pgrid(8, 4);    // 1024/128 x 512/128

    // ---- encoder: 63 steps x 3 layers ----
    for (int t = 0; t < TENC; ++t) {
        const int p = t & 1, wp = p ^ 1;
        for (int l = 0; l < NLAY; ++l) {
            const __hip_bfloat16* ax = (l == 0)
                ? Xbf + (size_t)t * BHE
                : Hbuf + ((size_t)(l - 1) * 2 + wp) * BHE;
            const __hip_bfloat16* ah = Hbuf + ((size_t)l * 2 + p) * BHE;
            lstm_cell_kernel<<<cgrid, 256, 0, stream>>>(
                ax, ah, WcatE + (size_t)l * 2048 * 1024, bE + l * 2048,
                Cbuf + (size_t)l * BHE, Hbuf + ((size_t)l * 2 + wp) * BHE, nullptr);
        }
    }

    // ---- decoder: 64 steps x 3 layers + projection ----
    for (int s = 0; s < TDEC; ++s) {
        const int t = TENC + s, p = t & 1, wp = p ^ 1;
        for (int l = 0; l < NLAY; ++l) {
            const __hip_bfloat16* ax = (l == 0)
                ? ((s == 0) ? Xbf + (size_t)(TENC - 1) * BHE : Din + (size_t)p * BHE)
                : Hbuf + ((size_t)(l - 1) * 2 + wp) * BHE;
            const __hip_bfloat16* ah = Hbuf + ((size_t)l * 2 + p) * BHE;
            __hip_bfloat16* hd1 = (l == NLAY - 1) ? Din + (size_t)wp * BHE : nullptr;
            lstm_cell_kernel<<<cgrid, 256, 0, stream>>>(
                ax, ah, WcatD + (size_t)l * 2048 * 1024, bD + l * 2048,
                Cbuf + (size_t)l * BHE, Hbuf + ((size_t)l * 2 + wp) * BHE, hd1);
        }
        proj_kernel<<<pgrid, 256, 0, stream>>>(Din + (size_t)wp * BHE, OutWbf, oB, out, s);
    }
}
